// HGCN_62130996904263
// MI455X (gfx1250) — matrix operand order repaired and checked by reading, not compiled, not run on hardware
//
#include <hip/hip_runtime.h>

// ---------------------------------------------------------------------------
// HGCN forward for MI455X (gfx1250, wave32).
// GEMMs: v_wmma_f32_16x16x32_bf16, B staged to LDS via TDM tensor_load_to_lds
// (s_wait_tensorcnt), A streamed once per row-strip. Elementwise math is
// wave-per-row with __shfl_xor reductions; aggregation is wave-per-edge
// global_atomic_add_f32.
// ---------------------------------------------------------------------------

typedef __attribute__((ext_vector_type(16))) __bf16 v16bf;
typedef __attribute__((ext_vector_type(8)))  float  v8f;
typedef unsigned int u32x4 __attribute__((ext_vector_type(4)));
typedef int          i32x4 __attribute__((ext_vector_type(4)));
typedef int          i32x8 __attribute__((ext_vector_type(8)));

#if __has_builtin(__builtin_amdgcn_tensor_load_to_lds)
#define HAVE_TDM 1
#endif

static constexpr float kMinNorm = 1e-15f;
static constexpr float kEps     = 1e-7f;
static constexpr float kMaxNorm = 1e6f;

// LDS image of B has a 272-byte row pitch (128 bf16 + 8 bf16 pad) so that
// lane n (reading row n) hits LDS bank 4n mod 64 -> conflict-free.
static constexpr int kBPitch = 136;   // elements (bf16)

static __device__ __forceinline__ float wsum(float v) {
#pragma unroll
  for (int off = 16; off > 0; off >>= 1) v += __shfl_xor(v, off, 32);
  return v;
}
static __device__ __forceinline__ float clip15(float t) {
  return fminf(fmaxf(t, -15.f), 15.f);
}
static __device__ __forceinline__ float arcosh_clipped(float x) {
  x = fmaxf(x, 1.f + kEps);
  return logf(x + sqrtf(x * x - 1.f));
}
static __device__ __forceinline__ unsigned short f2bf(float f) {
  unsigned int u = __float_as_uint(f);
  u += 0x7FFFu + ((u >> 16) & 1u);   // round-to-nearest-even
  return (unsigned short)(u >> 16);
}

// ---------------------------------------------------------------------------
// Stage 0: h = [0, x]; H = proj(expmap0(proj_tan0(h), C0), C0).  Wave per row.
// ---------------------------------------------------------------------------
__global__ __launch_bounds__(256) void embed_kernel(
    const float* __restrict__ x, float* __restrict__ H, int N) {
  const int row  = blockIdx.x * (blockDim.x >> 5) + (threadIdx.x >> 5);
  const int lane = threadIdx.x & 31;
  if (row >= N) return;
  const float K = 3.f, sqrtK = sqrtf(3.f);   // c = 1/3
  float v[4]; float ss = 0.f;
#pragma unroll
  for (int i = 0; i < 4; ++i) {
    int f = lane + 32 * i;
    float val = (f == 0) ? 0.f : x[(size_t)row * 127 + (f - 1)];
    v[i] = val; ss += val * val;
  }
  ss = wsum(ss);
  float yn = fmaxf(sqrtf(ss), kMinNorm);
  float sc = sqrtK * sinhf(clip15(yn / sqrtK)) / yn;
  float y1[4]; float ss2 = 0.f;
#pragma unroll
  for (int i = 0; i < 4; ++i) {
    int f = lane + 32 * i;
    y1[i] = (f == 0) ? 0.f : v[i] * sc;
    ss2 += y1[i] * y1[i];
  }
  ss2 = wsum(ss2);
  float x0 = sqrtf(fmaxf(K + ss2, kEps));
#pragma unroll
  for (int i = 0; i < 4; ++i) {
    int f = lane + 32 * i;
    H[(size_t)row * 128 + f] = (f == 0) ? x0 : y1[i];
  }
}

// ---------------------------------------------------------------------------
// logmap0 of manifold rows -> bf16 GEMM operand A (N x 128).  Wave per row.
// ---------------------------------------------------------------------------
__global__ __launch_bounds__(256) void logmap_bf16_kernel(
    const float* __restrict__ H, unsigned short* __restrict__ A, float c, int N) {
  const int row  = blockIdx.x * (blockDim.x >> 5) + (threadIdx.x >> 5);
  const int lane = threadIdx.x & 31;
  if (row >= N) return;
  const float K = 1.f / c, sqrtK = sqrtf(K);
  const float* hrow = H + (size_t)row * 128;
  float h0 = hrow[0];
  float y[4]; float ss = 0.f;
#pragma unroll
  for (int i = 0; i < 4; ++i) {
    int f = lane + 32 * i;
    float v = (f == 0) ? 0.f : hrow[f];
    y[i] = v; ss += v * v;
  }
  ss = wsum(ss);
  float yn = fmaxf(sqrtf(ss), kMinNorm);
  float sc = sqrtK * arcosh_clipped(h0 / sqrtK) / yn;
#pragma unroll
  for (int i = 0; i < 4; ++i) {
    int f = lane + 32 * i;
    A[(size_t)row * 128 + f] = f2bf((f == 0) ? 0.f : y[i] * sc);
  }
}

// ---------------------------------------------------------------------------
// fp32 -> bf16 weight conversion.
// ---------------------------------------------------------------------------
__global__ void w2bf_kernel(const float* __restrict__ s,
                            unsigned short* __restrict__ d, int n) {
  int i = blockIdx.x * blockDim.x + threadIdx.x;
  if (i < n) d[i] = f2bf(s[i]);
}

// ---------------------------------------------------------------------------
// Hyperbolic bias: uv = logmap0(proj(expmap0(proj_tan0(b), c), c), c).
// One wave.
// ---------------------------------------------------------------------------
template <int D>
__global__ void hb_kernel(const float* __restrict__ b, float c,
                          float* __restrict__ uv) {
  constexpr int IT = D / 32;
  const int lane = threadIdx.x;
  const float K = 1.f / c, sqrtK = sqrtf(K);
  float bt[IT]; float ss = 0.f;
#pragma unroll
  for (int i = 0; i < IT; ++i) {
    int f = lane + 32 * i;
    float v = (f == 0) ? 0.f : b[f];
    bt[i] = v; ss += v * v;
  }
  ss = wsum(ss);
  float yn = fmaxf(sqrtf(ss), kMinNorm);
  float sc = sqrtK * sinhf(clip15(yn / sqrtK)) / yn;
  float y[IT]; float ss2 = 0.f;
#pragma unroll
  for (int i = 0; i < IT; ++i) {
    int f = lane + 32 * i;
    y[i] = (f == 0) ? 0.f : bt[i] * sc;
    ss2 += y[i] * y[i];
  }
  ss2 = wsum(ss2);
  float hb0 = sqrtf(fmaxf(K + ss2, kEps));
  float yn2 = fmaxf(sqrtf(ss2), kMinNorm);
  float sc2 = sqrtK * arcosh_clipped(hb0 / sqrtK) / yn2;
#pragma unroll
  for (int i = 0; i < IT; ++i) {
    int f = lane + 32 * i;
    uv[f] = (f == 0) ? 0.f : y[i] * sc2;
  }
}

// ---------------------------------------------------------------------------
// WMMA GEMM: C(N x OUTD) = A(N x 128, bf16) @ W(OUTD x 128, bf16)^T.
// Block: stage B into LDS (TDM tensor_load_to_lds with row padding), then
// each wave computes a full 16-row x OUTD strip: A fragment loaded once per
// K-chunk, B fragments from LDS, TN accumulators.
// ---------------------------------------------------------------------------
template <int OUTD>
__global__ __launch_bounds__(256) void gemm_wmma_kernel(
    const unsigned short* __restrict__ A, const unsigned short* __restrict__ W,
    float* __restrict__ C, int N) {
  constexpr int TN = OUTD / 16;
  __shared__ unsigned short Bs[OUTD * kBPitch];

  // ---- Stage B (OUTD x 128 bf16) into LDS with 272B row pitch ----
#if defined(HAVE_TDM)
  if (threadIdx.x == 0) {
    unsigned long long ga = (unsigned long long)(size_t)W;
    unsigned ldsoff = (unsigned)(size_t)(&Bs[0]);   // flat low 32b == LDS offset
    // D# group 0: count=1, lds_addr, 57b global_addr, type=2 ("image")
    u32x4 g0 = { 1u,
                 ldsoff,
                 (unsigned)(ga & 0xFFFFFFFFull),
                 (unsigned)((ga >> 32) & 0x01FFFFFFull) | (2u << 30) };
    // D# group 1: data_size=2B (code 1); pad_enable, pad_interval=64 dwords
    // (code 5), pad_amount=4 dwords (code 3) -> row pitch 272B in LDS.
    // tensor_dim0=128, tensor_dim1=OUTD, tile_dim0=128, tile_dim1=OUTD,
    // tensor_dim0_stride=128.
    i32x8 g1 = { (int)((1u << 16) | (1u << 20) | (5u << 22) | (3u << 25)),
                 (int)(128u << 16),                 // tensor_dim0[15:0] @ bits 63:48
                 (int)((unsigned)OUTD << 16),       // tdim0 hi16=0 | tensor_dim1 lo16
                 (int)(128u << 16),                 // tdim1 hi16=0 | tile_dim0=128
                 (int)OUTD,                         // tile_dim1=OUTD, tile_dim2=0
                 128,                               // tensor_dim0_stride lo32
                 0, 0 };                            // stride0 hi, stride1
    i32x4 gz4 = {0, 0, 0, 0};
#if defined(__clang_major__) && (__clang_major__ >= 23)
    i32x8 gz8 = {0, 0, 0, 0, 0, 0, 0, 0};
    __builtin_amdgcn_tensor_load_to_lds(g0, g1, gz4, gz4, gz8, 0);
#else
    __builtin_amdgcn_tensor_load_to_lds(g0, g1, gz4, gz4, 0);
#endif
#if __has_builtin(__builtin_amdgcn_s_wait_tensorcnt)
    __builtin_amdgcn_s_wait_tensorcnt((short)0);
#endif
  }
#else
  {
    // Fallback: cooperative 8B copies with the same padded pitch.
    for (int i = threadIdx.x; i < OUTD * 32; i += 256) {
      int r = i >> 5, c4 = i & 31;               // c4: 4-element chunk
      *(unsigned long long*)(&Bs[r * kBPitch + c4 * 4]) =
          *(const unsigned long long*)(W + r * 128 + c4 * 4);
    }
  }
#endif
  __syncthreads();

  const int lane   = threadIdx.x & 31;
  const int tm     = blockIdx.x * 8 + (threadIdx.x >> 5);
  const int tilesM = (N + 15) >> 4;
  if (tm >= tilesM) return;
  const int row0 = tm << 4;
  const int half = lane >> 4;
  const int rowA = min(row0 + (lane & 15), N - 1);
  const unsigned short* arow = A + (size_t)rowA * 128;

  union F { v16bf v; unsigned int u[8]; };
  v8f acc[TN];
  const v8f vz = {0.f, 0.f, 0.f, 0.f, 0.f, 0.f, 0.f, 0.f};
#pragma unroll
  for (int t = 0; t < TN; ++t) acc[t] = vz;

#pragma unroll
  for (int kc = 0; kc < 128; kc += 32) {
    F a;
#pragma unroll
    for (int p = 0; p < 8; ++p) {   // A: VGPR p holds K pair {2p | 8+2p} + 8*half
      int kbase = ((p < 4) ? (2 * p) : (8 + 2 * p)) + 8 * half;
      a.u[p] = *(const unsigned int*)(arow + kc + kbase);
    }
#pragma unroll
    for (int t = 0; t < TN; ++t) {
      F b;                           // B: VGPR j holds K pair {2j | 8+2j} + 8*half
      const unsigned short* brow =
          &Bs[(t * 16 + (lane & 15)) * kBPitch + kc];
#pragma unroll
      for (int j = 0; j < 8; ++j) b.u[j] = *(const unsigned int*)(brow + ((j < 4) ? (2 * j) : (8 + 2 * j)) + 8 * half);
      acc[t] = __builtin_amdgcn_wmma_f32_16x16x32_bf16(
          false, a.v, false, b.v, (short)0, acc[t], false, false);
    }
  }

  // D layout: VGPR r -> M = r + 8*half, N = lane&15 (per 16x16 tile).
  if (row0 + 16 <= N) {
#pragma unroll
    for (int t = 0; t < TN; ++t) {
      float* crow = C + (size_t)(row0 + 8 * half) * OUTD + t * 16 + (lane & 15);
#pragma unroll
      for (int r = 0; r < 8; ++r) crow[(size_t)r * OUTD] = acc[t][r];
    }
  } else {
#pragma unroll
    for (int t = 0; t < TN; ++t) {
#pragma unroll
      for (int r = 0; r < 8; ++r) {
        int rr = row0 + r + 8 * half;
        if (rr < N) C[(size_t)rr * OUTD + t * 16 + (lane & 15)] = acc[t][r];
      }
    }
  }
}

// ---------------------------------------------------------------------------
// hyp_linear tail, fused:
//   res = proj(expmap0(mvt, c), c);  out = proj(expmap(ptransp0(res,u), res), c)
//   optionally followed by logmap0 (feeding hyp_agg).  Wave per row.
// ---------------------------------------------------------------------------
template <int D, bool OUT_LOG>
__global__ __launch_bounds__(256) void postlinear_kernel(
    const float* __restrict__ Mvt, const float* __restrict__ uvec, float c,
    float* __restrict__ out, int N) {
  constexpr int IT = D / 32;
  const int row  = blockIdx.x * (blockDim.x >> 5) + (threadIdx.x >> 5);
  const int lane = threadIdx.x & 31;
  if (row >= N) return;
  const float K = 1.f / c, sqrtK = sqrtf(K);
  const float* mrow = Mvt + (size_t)row * D;

  // expmap0(mvt, c)
  float mv[IT]; float ss = 0.f;
#pragma unroll
  for (int i = 0; i < IT; ++i) {
    int f = lane + 32 * i;
    float v = (f == 0) ? 0.f : mrow[f];
    mv[i] = v; ss += v * v;
  }
  ss = wsum(ss);
  float yn = fmaxf(sqrtf(ss), kMinNorm);
  float sc = sqrtK * sinhf(clip15(yn / sqrtK)) / yn;
  float y[IT]; float ss2 = 0.f;
#pragma unroll
  for (int i = 0; i < IT; ++i) {
    int f = lane + 32 * i;
    y[i] = (f == 0) ? 0.f : mv[i] * sc;
    ss2 += y[i] * y[i];
  }
  ss2 = wsum(ss2);
  float res0  = sqrtf(fmaxf(K + ss2, kEps));      // proj
  float ynres = fmaxf(sqrtf(ss2), kMinNorm);

  // ptransp0(res, u): alpha = dot(yhat, u[1:]) / sqrtK ; t = u - alpha*v
  float uval[IT]; float du = 0.f;
#pragma unroll
  for (int i = 0; i < IT; ++i) {
    int f = lane + 32 * i;
    uval[i] = uvec[f];
    if (f > 0) du += (y[i] / ynres) * uval[i];
  }
  du = wsum(du);
  float alpha = du / sqrtK;
  float coef  = alpha * (sqrtK - res0) / ynres;   // alpha * (sqrtK - x0) / yn
  float t[IT];
#pragma unroll
  for (int i = 0; i < IT; ++i) {
    int f = lane + 32 * i;
    t[i] = (f == 0) ? 0.f : (uval[i] - coef * y[i]);
  }
  // proj_tan(t, res): t0 = dot(res[1:], t[1:]) / clip(res0)
  float ux = 0.f;
#pragma unroll
  for (int i = 0; i < IT; ++i) {
    int f = lane + 32 * i;
    if (f > 0) ux += y[i] * t[i];
  }
  ux = wsum(ux);
  float t0 = ux / fmaxf(res0, kEps);
#pragma unroll
  for (int i = 0; i < IT; ++i) {
    int f = lane + 32 * i;
    if (f == 0) t[i] = t0;
  }

  // expmap(t, res, c)
  float sv = 0.f;
#pragma unroll
  for (int i = 0; i < IT; ++i) sv += t[i] * t[i];
  sv = wsum(sv);
  float mdot  = sv - 2.f * t0 * t0;               // minkowski_dot(t,t)
  float normu = fminf(sqrtf(fmaxf(mdot, kEps)), kMaxNorm);
  float th2   = fmaxf(normu / sqrtK, kMinNorm);
  float ch    = coshf(fminf(th2, 15.f));
  float sh    = sinhf(fminf(th2, 15.f)) / th2;
  float r2[IT]; float ss3 = 0.f;
#pragma unroll
  for (int i = 0; i < IT; ++i) {
    int f = lane + 32 * i;
    float xc = (f == 0) ? res0 : y[i];
    r2[i] = ch * xc + sh * t[i];
    if (f > 0) ss3 += r2[i] * r2[i];
  }
  ss3 = wsum(ss3);
  float o0 = sqrtf(fmaxf(K + ss3, kEps));         // proj
  float* orow = out + (size_t)row * D;
  if (OUT_LOG) {                                  // logmap0 for hyp_agg
    float yn3 = fmaxf(sqrtf(ss3), kMinNorm);
    float sc3 = sqrtK * arcosh_clipped(o0 / sqrtK) / yn3;
#pragma unroll
    for (int i = 0; i < IT; ++i) {
      int f = lane + 32 * i;
      orow[f] = (f == 0) ? 0.f : r2[i] * sc3;
    }
  } else {
#pragma unroll
    for (int i = 0; i < IT; ++i) {
      int f = lane + 32 * i;
      orow[f] = (f == 0) ? o0 : r2[i];
    }
  }
}

// ---------------------------------------------------------------------------
// Zero init for the segment-sum accumulator.
// ---------------------------------------------------------------------------
__global__ void zero_kernel(float* __restrict__ p, size_t n) {
  size_t i = (size_t)blockIdx.x * blockDim.x + threadIdx.x;
  size_t stride = (size_t)gridDim.x * blockDim.x;
  for (; i < n; i += stride) p[i] = 0.f;
}

// ---------------------------------------------------------------------------
// Edge scatter: Agg[dst] += Xt[src] * w.  One wave per edge, 128 features.
// ---------------------------------------------------------------------------
__global__ __launch_bounds__(256) void scatter_kernel(
    const float* __restrict__ Xt, const int* __restrict__ src,
    const int* __restrict__ dst, const float* __restrict__ w,
    float* __restrict__ Agg, int E) {
  const int e    = blockIdx.x * (blockDim.x >> 5) + (threadIdx.x >> 5);
  const int lane = threadIdx.x & 31;
  if (e >= E) return;
  const int s = src[e], d = dst[e];
  const float wt = w[e];
  const float* xs = Xt + (size_t)s * 128;
  float* ad = Agg + (size_t)d * 128;
#pragma unroll
  for (int i = 0; i < 4; ++i) {
    int f = lane + 32 * i;
    atomicAdd(&ad[f], xs[f] * wt);
  }
}

// ---------------------------------------------------------------------------
// Fused: H = proj(expmap0(Agg, c_in), c_in) -> relu(logmap0(., c_in))
//        -> proj(expmap0(proj_tan0(.), c_out), c_out).  Wave per row.
// ---------------------------------------------------------------------------
__global__ __launch_bounds__(256) void agg_act_kernel(
    const float* __restrict__ Agg, float cin, float cout,
    float* __restrict__ H, int N) {
  const int row  = blockIdx.x * (blockDim.x >> 5) + (threadIdx.x >> 5);
  const int lane = threadIdx.x & 31;
  if (row >= N) return;
  const float Ki = 1.f / cin,  sKi = sqrtf(Ki);
  const float Ko = 1.f / cout, sKo = sqrtf(Ko);
  const float* arow = Agg + (size_t)row * 128;

  float a[4]; float ss = 0.f;
#pragma unroll
  for (int i = 0; i < 4; ++i) {
    int f = lane + 32 * i;
    float v = (f == 0) ? 0.f : arow[f];
    a[i] = v; ss += v * v;
  }
  ss = wsum(ss);
  float yn = fmaxf(sqrtf(ss), kMinNorm);
  float sc = sKi * sinhf(clip15(yn / sKi)) / yn;          // expmap0(c_in)
  float y1[4]; float ss2 = 0.f;
#pragma unroll
  for (int i = 0; i < 4; ++i) {
    int f = lane + 32 * i;
    y1[i] = (f == 0) ? 0.f : a[i] * sc;
    ss2 += y1[i] * y1[i];
  }
  ss2 = wsum(ss2);
  float x0  = sqrtf(fmaxf(Ki + ss2, kEps));               // proj(c_in)
  float yn1 = fmaxf(sqrtf(ss2), kMinNorm);
  float sc2 = sKi * arcosh_clipped(x0 / sKi) / yn1;       // logmap0(c_in)
  float lt[4]; float ss3 = 0.f;
#pragma unroll
  for (int i = 0; i < 4; ++i) {
    int f = lane + 32 * i;
    lt[i] = (f == 0) ? 0.f : fmaxf(y1[i] * sc2, 0.f);     // relu + proj_tan0
    ss3 += lt[i] * lt[i];
  }
  ss3 = wsum(ss3);
  float yn2 = fmaxf(sqrtf(ss3), kMinNorm);
  float sc3 = sKo * sinhf(clip15(yn2 / sKo)) / yn2;       // expmap0(c_out)
  float y3[4]; float ss4 = 0.f;
#pragma unroll
  for (int i = 0; i < 4; ++i) {
    int f = lane + 32 * i;
    y3[i] = (f == 0) ? 0.f : lt[i] * sc3;
    ss4 += y3[i] * y3[i];
  }
  ss4 = wsum(ss4);
  float o0 = sqrtf(fmaxf(Ko + ss4, kEps));                // proj(c_out)
#pragma unroll
  for (int i = 0; i < 4; ++i) {
    int f = lane + 32 * i;
    H[(size_t)row * 128 + f] = (f == 0) ? o0 : y3[i];
  }
}

// ---------------------------------------------------------------------------
// Host-side orchestration (graph-capture safe: only stream launches).
// ---------------------------------------------------------------------------
extern "C" void kernel_launch(void* const* d_in, const int* in_sizes, int n_in,
                              void* d_out, int out_size, void* d_ws,
                              size_t ws_size, hipStream_t stream) {
  (void)n_in; (void)out_size; (void)ws_size;
  const float* x  = (const float*)d_in[0];
  const int*   ei = (const int*)  d_in[1];
  const float* ew = (const float*)d_in[2];
  const float* W1 = (const float*)d_in[3];
  const float* b1 = (const float*)d_in[4];
  const float* W2 = (const float*)d_in[5];
  const float* b2 = (const float*)d_in[6];
  const float* Wl = (const float*)d_in[7];
  const float* bl = (const float*)d_in[8];

  const int N = in_sizes[0] / 127;
  const int E = in_sizes[2];
  const float C0 = 1.f / 3.f, C1 = 0.5f, C2 = 1.f;

  const size_t N128 = (size_t)N * 128;
  float* H   = (float*)d_ws;           // N x 128 manifold points
  float* Mvt = H   + N128;             // N x 128 GEMM output (tangent)
  float* Xt2 = Mvt + N128;             // N x 128 scatter source (tangent)
  float* Agg = Xt2 + N128;             // N x 128 segment-sum accumulator
  unsigned short* Abf = (unsigned short*)(Agg + N128);   // N x 128 bf16 A
  unsigned short* Wbf = Abf + N128;                      // <=128x128 bf16 W
  float* uv = (float*)(Wbf + 128 * 128);                 // 128 bias tangent

  const int* src = ei;
  const int* dst = ei + E;

  dim3 blk(256);
  const int rowBlocks  = (N + 7) / 8;
  const int edgeBlocks = (E + 7) / 8;
  const int tilesM     = (N + 15) / 16;
  const int gemmBlocks = (tilesM + 7) / 8;   // 8 row-strips per block

  // Stage 0: embed onto hyperboloid (c = C0)
  embed_kernel<<<rowBlocks, blk, 0, stream>>>(x, H, N);

  // ---- Layer 1 (c_in = C0, act -> C1) ----
  w2bf_kernel<<<(128 * 128 + 255) / 256, blk, 0, stream>>>(W1, Wbf, 128 * 128);
  hb_kernel<128><<<1, 32, 0, stream>>>(b1, C0, uv);
  logmap_bf16_kernel<<<rowBlocks, blk, 0, stream>>>(H, Abf, C0, N);
  gemm_wmma_kernel<128><<<gemmBlocks, blk, 0, stream>>>(Abf, Wbf, Mvt, N);
  postlinear_kernel<128, true><<<rowBlocks, blk, 0, stream>>>(Mvt, uv, C0, Xt2, N);
  zero_kernel<<<2048, blk, 0, stream>>>(Agg, N128);
  scatter_kernel<<<edgeBlocks, blk, 0, stream>>>(Xt2, src, dst, ew, Agg, E);
  agg_act_kernel<<<rowBlocks, blk, 0, stream>>>(Agg, C0, C1, H, N);

  // ---- Layer 2 (c_in = C1, act -> C2) ----
  w2bf_kernel<<<(128 * 128 + 255) / 256, blk, 0, stream>>>(W2, Wbf, 128 * 128);
  hb_kernel<128><<<1, 32, 0, stream>>>(b2, C1, uv);
  logmap_bf16_kernel<<<rowBlocks, blk, 0, stream>>>(H, Abf, C1, N);
  gemm_wmma_kernel<128><<<gemmBlocks, blk, 0, stream>>>(Abf, Wbf, Mvt, N);
  postlinear_kernel<128, true><<<rowBlocks, blk, 0, stream>>>(Mvt, uv, C1, Xt2, N);
  zero_kernel<<<2048, blk, 0, stream>>>(Agg, N128);
  scatter_kernel<<<edgeBlocks, blk, 0, stream>>>(Xt2, src, dst, ew, Agg, E);
  agg_act_kernel<<<rowBlocks, blk, 0, stream>>>(Agg, C1, C2, H, N);

  // ---- Final hyp_linear (c = C2, OUTD = 64) ----
  w2bf_kernel<<<(64 * 128 + 255) / 256, blk, 0, stream>>>(Wl, Wbf, 64 * 128);
  hb_kernel<64><<<1, 32, 0, stream>>>(bl, C2, uv);
  logmap_bf16_kernel<<<rowBlocks, blk, 0, stream>>>(H, Abf, C2, N);
  gemm_wmma_kernel<64><<<gemmBlocks, blk, 0, stream>>>(Abf, Wbf, Mvt, N);
  postlinear_kernel<64, false><<<rowBlocks, blk, 0, stream>>>(
      Mvt, uv, C2, (float*)d_out, N);
}
